// GATv2Conv_19000935317881
// MI455X (gfx1250) — matrix operand order repaired and checked by reading, not compiled, not run on hardware
//
#include <hip/hip_runtime.h>
#include <hip/hip_bf16.h>

typedef __attribute__((ext_vector_type(16))) _Float16 v16h;
typedef __attribute__((ext_vector_type(8)))  float    v8f;

#define N_NODES 50000
#define N_EDGES 800000
// concat row for h_a: [src(0..63) | edge(64..95) | tgt(96..159)], K = 160
// h_m input = K range [64,160) of the same row, K' = K-64, 96 wide

// ---- workspace layout (bytes) ----
#define WS_WTA    0            // f16 [128][160]  = 40960 B
#define WS_WTM    40960        // f16 [128][96]   = 24576 B
#define WS_NMAX   65536        // u32 [N*4]       = 800000 B  (encoded max; later reused as f32 max)
#define WS_DENOM  865536       // f32 [N*4]       = 800000 B
#define WS_DEAD   1665536      // i32 [N]         = 200000 B
#define WS_LOGIT  1865536      // f32 [E*4]       = 12.8 MB  (logits, then u = exp(...))

__device__ __forceinline__ unsigned encf(float f) {
    unsigned b = __float_as_uint(f);
    return (b & 0x80000000u) ? ~b : (b | 0x80000000u);
}
__device__ __forceinline__ float decf(unsigned u) {
    unsigned b = (u & 0x80000000u) ? (u & 0x7FFFFFFFu) : ~u;
    return __uint_as_float(b);
}

// pointer into the virtual concat row [src|edge|tgt] at column K (chunk of 8 never straddles)
__device__ __forceinline__ const float* rowPtr(const float* nodef, const float* edgef,
                                               int me, int s, int t, int K) {
    if (K < 64)  return nodef + (size_t)s * 64 + K;
    if (K < 96)  return edgef + (size_t)me * 32 + (K - 64);
    return nodef + (size_t)t * 64 + (K - 96);
}

// Build one 16-bit A fragment (16x32 MxK tile starting at global column kbase).
// Lane l: row m = l&15; halfs j=0..15 -> K = kbase + 16*hi + j
__device__ __forceinline__ v16h load_afrag(const float* nodef, const float* edgef,
                                           int me, int s, int t, int kbase, int hi) {
    const float4* p0 = (const float4*)rowPtr(nodef, edgef, me, s, t, kbase + 16 * hi);
    const float4* p1 = (const float4*)rowPtr(nodef, edgef, me, s, t, kbase + 16 * hi + 8);
    float4 x0 = p0[0], x1 = p0[1], y0 = p1[0], y1 = p1[1];
    v16h a;
    a[0]  = (_Float16)x0.x; a[1]  = (_Float16)x0.y; a[2]  = (_Float16)x0.z; a[3]  = (_Float16)x0.w;
    a[4]  = (_Float16)x1.x; a[5]  = (_Float16)x1.y; a[6]  = (_Float16)x1.z; a[7]  = (_Float16)x1.w;
    a[8]  = (_Float16)y0.x; a[9]  = (_Float16)y0.y; a[10] = (_Float16)y0.z; a[11] = (_Float16)y0.w;
    a[12] = (_Float16)y1.x; a[13] = (_Float16)y1.y; a[14] = (_Float16)y1.z; a[15] = (_Float16)y1.w;
    return a;
}

#define WMMA(a, b, c) __builtin_amdgcn_wmma_f32_16x16x32_f16(false, (a), false, (b), (short)0, (c), false, false)

// ---------------- K0: init + weight transpose/convert ----------------
__global__ __launch_bounds__(256) void k0_init(float* dout, _Float16* wta, _Float16* wtm,
                                               unsigned* nmax, float* denom,
                                               const float* Wa, const float* Wm) {
    int i = blockIdx.x * 256 + threadIdx.x;
    if (i < N_NODES * 128) dout[i] = 0.0f;
    if (i < 128 * 160) { int n = i / 160, k = i % 160; wta[i] = (_Float16)Wa[(size_t)k * 128 + n]; }
    if (i < 128 * 96)  { int n = i / 96,  k = i % 96;  wtm[i] = (_Float16)Wm[(size_t)k * 128 + n]; }
    if (i < N_NODES * 4) { nmax[i] = 0u; denom[i] = 0.0f; }
}

// ---------------- K1: h_a GEMM (2 M-tiles/wave) + leakyrelu . a_w + masked logits + seg max ----
__global__ __launch_bounds__(256) void k1_edge_attn(const float* nodef, const float* edgef,
                                                    const int* ei, const unsigned char* mask,
                                                    const float* aw, const _Float16* wta,
                                                    float* wsLogit, unsigned* nmax) {
    int lane  = threadIdx.x & 31;
    int wtile = (blockIdx.x * 256 + threadIdx.x) >> 5;   // 32 edges (two 16-row M-tiles) per wave
    if (wtile * 32 >= N_EDGES) return;
    int hi = lane >> 4, n = lane & 15;

    int me0 = wtile * 32 + n;            // A-side rows owned by this lane
    int me1 = me0 + 16;
    int sA0 = ei[me0], tA0 = ei[N_EDGES + me0];
    int sA1 = ei[me1], tA1 = ei[N_EDGES + me1];

    v16h af0[5], af1[5];
#pragma unroll
    for (int kt = 0; kt < 5; ++kt) {
        af0[kt] = load_afrag(nodef, edgef, me0, sA0, tA0, 32 * kt, hi);
        af1[kt] = load_afrag(nodef, edgef, me1, sA1, tA1, 32 * kt, hi);
    }

    float aw0 = aw[n], aw1 = aw[16 + n];

#pragma unroll
    for (int h = 0; h < 4; ++h) {
        v8f c00 = {0.f,0.f,0.f,0.f,0.f,0.f,0.f,0.f};   // M-tile0, N-tile 2h
        v8f c01 = {0.f,0.f,0.f,0.f,0.f,0.f,0.f,0.f};   // M-tile0, N-tile 2h+1
        v8f c10 = {0.f,0.f,0.f,0.f,0.f,0.f,0.f,0.f};   // M-tile1, N-tile 2h
        v8f c11 = {0.f,0.f,0.f,0.f,0.f,0.f,0.f,0.f};   // M-tile1, N-tile 2h+1
#pragma unroll
        for (int kt = 0; kt < 5; ++kt) {
            v16h b0 = *(const v16h*)(wta + (size_t)(32 * h + n) * 160 + 32 * kt + 16 * hi);
            c00 = WMMA(af0[kt], b0, c00);
            c10 = WMMA(af1[kt], b0, c10);
            v16h b1 = *(const v16h*)(wta + (size_t)(32 * h + 16 + n) * 160 + 32 * kt + 16 * hi);
            c01 = WMMA(af0[kt], b1, c01);
            c11 = WMMA(af1[kt], b1, c11);
        }
        float p0[8], p1[8];
#pragma unroll
        for (int r = 0; r < 8; ++r) {
            float x0 = c00[r]; x0 = x0 > 0.f ? x0 : 0.2f * x0;
            float x1 = c01[r]; x1 = x1 > 0.f ? x1 : 0.2f * x1;
            p0[r] = x0 * aw0 + x1 * aw1;
            float y0 = c10[r]; y0 = y0 > 0.f ? y0 : 0.2f * y0;
            float y1 = c11[r]; y1 = y1 > 0.f ? y1 : 0.2f * y1;
            p1[r] = y0 * aw0 + y1 * aw1;
        }
#pragma unroll
        for (int r = 0; r < 8; ++r) {
            p0[r] += __shfl_xor(p0[r], 1, 32);
            p0[r] += __shfl_xor(p0[r], 2, 32);
            p0[r] += __shfl_xor(p0[r], 4, 32);
            p0[r] += __shfl_xor(p0[r], 8, 32);
            p1[r] += __shfl_xor(p1[r], 1, 32);
            p1[r] += __shfl_xor(p1[r], 2, 32);
            p1[r] += __shfl_xor(p1[r], 4, 32);
            p1[r] += __shfl_xor(p1[r], 8, 32);
        }
        if (n == h) {
#pragma unroll
            for (int r = 0; r < 8; ++r) {
                int e0 = wtile * 32 + 8 * hi + r;
                float lg0 = mask[e0] ? p0[r] : -__builtin_inff();
                wsLogit[(size_t)e0 * 4 + h] = lg0;
                atomicMax(&nmax[(size_t)ei[e0] * 4 + h], encf(lg0));
                int e1 = e0 + 16;
                float lg1 = mask[e1] ? p1[r] : -__builtin_inff();
                wsLogit[(size_t)e1 * 4 + h] = lg1;
                atomicMax(&nmax[(size_t)ei[e1] * 4 + h], encf(lg1));
            }
        }
    }
}

// ---------------- K2: dead-node detection + decode maxima ----------------
__global__ __launch_bounds__(256) void k2_node(unsigned* nmax, int* dead) {
    int i = blockIdx.x * 256 + threadIdx.x;
    if (i >= N_NODES) return;
    unsigned u0 = nmax[(size_t)i * 4 + 0], u1 = nmax[(size_t)i * 4 + 1];
    unsigned u2 = nmax[(size_t)i * 4 + 2], u3 = nmax[(size_t)i * 4 + 3];
    int d = (u0 <= 0x007FFFFFu);    // max == -inf or no incident edges
    dead[i] = d;
    float* f = (float*)nmax;
    f[(size_t)i * 4 + 0] = d ? 0.f : decf(u0);
    f[(size_t)i * 4 + 1] = d ? 0.f : decf(u1);
    f[(size_t)i * 4 + 2] = d ? 0.f : decf(u2);
    f[(size_t)i * 4 + 3] = d ? 0.f : decf(u3);
}

// ---------------- K3: u = exp(l - max) (dead -> 1), accumulate denominators ----------------
__global__ __launch_bounds__(256) void k3_exp(const int* ei, const int* dead, const float* nmaxF,
                                              float* wsLogit, float* denom) {
    int i = blockIdx.x * 256 + threadIdx.x;
    if (i >= N_EDGES * 4) return;
    int e = i >> 2, h = i & 3;
    int s = ei[e];
    float u;
    if (dead[s]) u = 1.0f;
    else         u = __expf(wsLogit[i] - nmaxF[(size_t)s * 4 + h]);   // exp(-inf)=0 for masked
    wsLogit[i] = u;
    atomicAdd(&denom[(size_t)s * 4 + h], u);
}

// ---------------- K4: h_m GEMM (2 M-tiles/wave) + attn scaling + segment-sum scatter --------
__global__ __launch_bounds__(256) void k4_edge_msg(const float* nodef, const float* edgef,
                                                   const int* ei, const _Float16* wtm,
                                                   const float* wsU, const float* denom,
                                                   float* dout) {
    int lane  = threadIdx.x & 31;
    int wtile = (blockIdx.x * 256 + threadIdx.x) >> 5;
    if (wtile * 32 >= N_EDGES) return;
    int hi = lane >> 4, n = lane & 15;

    int me0 = wtile * 32 + n;
    int me1 = me0 + 16;
    int sA0 = ei[me0], tA0 = ei[N_EDGES + me0];
    int sA1 = ei[me1], tA1 = ei[N_EDGES + me1];

    v16h af0[3], af1[3];
#pragma unroll
    for (int kp = 0; kp < 3; ++kp) {
        af0[kp] = load_afrag(nodef, edgef, me0, sA0, tA0, 64 + 32 * kp, hi);  // K' = K-64 of h_m
        af1[kp] = load_afrag(nodef, edgef, me1, sA1, tA1, 64 + 32 * kp, hi);
    }

    int eR0[8], sR0[8], eR1[8], sR1[8];
#pragma unroll
    for (int r = 0; r < 8; ++r) {
        eR0[r] = wtile * 32 + 8 * hi + r;      sR0[r] = ei[eR0[r]];
        eR1[r] = eR0[r] + 16;                  sR1[r] = ei[eR1[r]];
    }

#pragma unroll
    for (int t = 0; t < 8; ++t) {
        v8f a0 = {0.f,0.f,0.f,0.f,0.f,0.f,0.f,0.f};
        v8f a1 = {0.f,0.f,0.f,0.f,0.f,0.f,0.f,0.f};
#pragma unroll
        for (int kp = 0; kp < 3; ++kp) {
            v16h b = *(const v16h*)(wtm + (size_t)(16 * t + n) * 96 + 32 * kp + 16 * hi);
            a0 = WMMA(af0[kp], b, a0);
            a1 = WMMA(af1[kp], b, a1);
        }
        int h = t >> 1;
#pragma unroll
        for (int r = 0; r < 8; ++r) {
            float u0 = wsU[(size_t)eR0[r] * 4 + h];
            float d0 = denom[(size_t)sR0[r] * 4 + h];
            atomicAdd(&dout[(size_t)sR0[r] * 128 + 16 * t + n], a0[r] * (u0 / d0));
            float u1 = wsU[(size_t)eR1[r] * 4 + h];
            float d1 = denom[(size_t)sR1[r] * 4 + h];
            atomicAdd(&dout[(size_t)sR1[r] * 128 + 16 * t + n], a1[r] * (u1 / d1));
        }
    }
}

extern "C" void kernel_launch(void* const* d_in, const int* in_sizes, int n_in,
                              void* d_out, int out_size, void* d_ws, size_t ws_size,
                              hipStream_t stream) {
    const float*         nodef = (const float*)d_in[0];
    const float*         edgef = (const float*)d_in[1];
    const int*           ei    = (const int*)d_in[2];
    const unsigned char* mask  = (const unsigned char*)d_in[3];
    const float*         Wa    = (const float*)d_in[4];
    const float*         Wm    = (const float*)d_in[5];
    const float*         aw    = (const float*)d_in[6];
    float*               dout  = (float*)d_out;

    char* ws = (char*)d_ws;
    _Float16* wta   = (_Float16*)(ws + WS_WTA);
    _Float16* wtm   = (_Float16*)(ws + WS_WTM);
    unsigned* nmax  = (unsigned*)(ws + WS_NMAX);
    float*    denom = (float*)(ws + WS_DENOM);
    int*      dead  = (int*)(ws + WS_DEAD);
    float*    wsLog = (float*)(ws + WS_LOGIT);

    (void)in_sizes; (void)n_in; (void)out_size; (void)ws_size;

    k0_init<<<(N_NODES * 128 + 255) / 256, 256, 0, stream>>>(dout, wta, wtm, nmax, denom, Wa, Wm);

    int nWaveTiles = N_EDGES / 32;              // 25000 wave-tiles (32 edges each)
    int gemmBlocks = (nWaveTiles + 7) / 8;      // 3125 blocks of 8 waves
    k1_edge_attn<<<gemmBlocks, 256, 0, stream>>>(nodef, edgef, ei, mask, aw, wta, wsLog, nmax);

    k2_node<<<(N_NODES + 255) / 256, 256, 0, stream>>>(nmax, dead);

    k3_exp<<<(N_EDGES * 4 + 255) / 256, 256, 0, stream>>>(ei, dead, (const float*)nmax, wsLog, denom);

    k4_edge_msg<<<gemmBlocks, 256, 0, stream>>>(nodef, edgef, ei, wtm, wsLog, denom, dout);
}
